// CognitiveNetwork_45896020525222
// MI455X (gfx1250) — compile-verified, matrix operand order checked by reading, not run on hardware
//
#include <hip/hip_runtime.h>
#include <hip/hip_bf16.h>

#define BSZ 2048
#define HD 1024
#define NCELLS 16
#define NACT 24
#define LNEPS 1e-5f
#define LDS_STRIDE 40   // shorts per LDS tile row (32 data + 8 pad), keeps 16B alignment
#define NKIT (HD / 32)

typedef __attribute__((ext_vector_type(16))) __bf16 v16bf;
typedef __attribute__((ext_vector_type(8)))  float  v8f;
typedef __attribute__((ext_vector_type(8)))  unsigned short usv8;
typedef __attribute__((ext_vector_type(4)))  unsigned short usv4;
typedef __attribute__((ext_vector_type(4)))  int iv4;

union FragU { v16bf v; usv8 h[2]; };

#if defined(__has_builtin)
#if __has_builtin(__builtin_amdgcn_global_load_async_to_lds_b128)
#define HAVE_ASYNC_LDS 1
#endif
#endif

__device__ __forceinline__ void cp16(unsigned short* lds_dst, const unsigned short* g_src) {
#ifdef HAVE_ASYNC_LDS
  __builtin_amdgcn_global_load_async_to_lds_b128(
      (__attribute__((address_space(1))) iv4*)g_src,
      (__attribute__((address_space(3))) iv4*)lds_dst, 0, 0);
#else
  *(usv8*)lds_dst = *(const usv8*)g_src;
#endif
}

__device__ __forceinline__ void wait_async() {
#ifdef HAVE_ASYNC_LDS
#if __has_builtin(__builtin_amdgcn_s_wait_asynccnt)
  __builtin_amdgcn_s_wait_asynccnt(0);
#else
  asm volatile("s_wait_asynccnt 0x0" ::: "memory");
#endif
#endif
}

__device__ __forceinline__ unsigned short f2bf(float f) {
  union { float f; unsigned int u; } v; v.f = f;
  unsigned int r = v.u + 0x7FFFu + ((v.u >> 16) & 1u);
  return (unsigned short)(r >> 16);
}

__device__ __forceinline__ float sigmoidf_(float x) {
  return 1.0f / (1.0f + __expf(-x));
}

// Load a 16x32 bf16 fragment from a row-major LDS tile (row stride LDS_STRIDE shorts).
// Lane l: M = l&15, Khalf = l>>4. Per-lane K runs: [8*kh, 8*kh+7] and [16+8*kh, 16+8*kh+7].
__device__ __forceinline__ v16bf ld_frag(const unsigned short* tile) {
  const int lane = threadIdx.x & 31;
  const int m  = lane & 15;
  const int kh = lane >> 4;
  const unsigned short* p = tile + m * LDS_STRIDE + kh * 8;
  FragU f;
  f.h[0] = *(const usv8*)(p);
  f.h[1] = *(const usv8*)(p + 16);
  return f.v;
}

__device__ __forceinline__ v8f wmma_bf16(v16bf a, v16bf b, v8f c) {
  return __builtin_amdgcn_wmma_f32_16x16x32_bf16(false, a, false, b, (short)0, c, false, false);
}

// ---------------------------------------------------------------- utility kernels
__global__ __launch_bounds__(256) void k_zero_f32(float* __restrict__ p, int n) {
  int i = 4 * (blockIdx.x * 256 + threadIdx.x);
  if (i + 3 < n) {
    float4 z; z.x = 0.f; z.y = 0.f; z.z = 0.f; z.w = 0.f;
    *(float4*)(p + i) = z;
  }
}

__global__ __launch_bounds__(256) void k_cvt_bf16(const float* __restrict__ in,
                                                  unsigned short* __restrict__ out, int n) {
  int i = 4 * (blockIdx.x * 256 + threadIdx.x);
  if (i + 3 < n) {
    float4 v = *(const float4*)(in + i);
    usv4 o;
    o.x = f2bf(v.x); o.y = f2bf(v.y); o.z = f2bf(v.z); o.w = f2bf(v.w);
    *(usv4*)(out + i) = o;
  }
}

// xi = bf16(x + 0.2*ctx)
__global__ __launch_bounds__(256) void k_xi(const float* __restrict__ x,
                                            const float* __restrict__ ctx,
                                            unsigned short* __restrict__ xi, int n) {
  int i = 4 * (blockIdx.x * 256 + threadIdx.x);
  if (i + 3 < n) {
    float4 xv = *(const float4*)(x + i);
    float4 cv = *(const float4*)(ctx + i);
    usv4 o;
    o.x = f2bf(xv.x + 0.2f * cv.x);
    o.y = f2bf(xv.y + 0.2f * cv.y);
    o.z = f2bf(xv.z + 0.2f * cv.z);
    o.w = f2bf(xv.w + 0.2f * cv.w);
    *(usv4*)(xi + i) = o;
  }
}

// ---------------------------------------------------------------- GEMM 1: perception
// p = relu(xi @ W_p[idx]^T + b_p[idx])   -> p_f32 [B,H]
__global__ __launch_bounds__(256) void k_gemm_percept(
    const unsigned short* __restrict__ xi,   // [B,H] bf16
    const unsigned short* __restrict__ Wp,   // [NC,H,H] bf16, rows are N, K contiguous
    const float* __restrict__ bp,            // [NC,H]
    const int* __restrict__ cidx, int t,
    float* __restrict__ pout) {
  const int idx = cidx[t];
  const unsigned short* W = Wp + (size_t)idx * HD * HD;
  const int m0 = blockIdx.y * 128;
  const int n0 = blockIdx.x * 128;
  __shared__ __align__(16) unsigned short sA[2][128 * LDS_STRIDE];
  __shared__ __align__(16) unsigned short sB[2][128 * LDS_STRIDE];
  const int tid = threadIdx.x;
  const int w = tid >> 5, lane = tid & 31;
  const int wm = w & 1, wn = w >> 1;   // 2x4 wave grid: 64 rows x 32 cols per wave
  v8f acc[4][2] = {};

  auto prefetch = [&](int buf, int k0) {
    #pragma unroll
    for (int c = tid; c < 512; c += 256) {
      int r = c >> 2, q = c & 3;
      cp16(&sA[buf][r * LDS_STRIDE + q * 8], xi + (size_t)(m0 + r) * HD + k0 + q * 8);
      cp16(&sB[buf][r * LDS_STRIDE + q * 8], W + (size_t)(n0 + r) * HD + k0 + q * 8);
    }
  };

  prefetch(0, 0);
  for (int it = 0; it < NKIT; ++it) {
    wait_async();
    __syncthreads();               // buf(it&1) ready; everyone done reading buf((it+1)&1)
    if (it + 1 < NKIT) prefetch((it + 1) & 1, (it + 1) * 32);
    const int cur = it & 1;
    v16bf bfr[2];
    #pragma unroll
    for (int nt = 0; nt < 2; nt++) bfr[nt] = ld_frag(&sB[cur][(wn * 32 + nt * 16) * LDS_STRIDE]);
    #pragma unroll
    for (int mt = 0; mt < 4; mt++) {
      v16bf a = ld_frag(&sA[cur][(wm * 64 + mt * 16) * LDS_STRIDE]);
      #pragma unroll
      for (int nt = 0; nt < 2; nt++) acc[mt][nt] = wmma_bf16(a, bfr[nt], acc[mt][nt]);
    }
  }

  const int kh = lane >> 4, nl = lane & 15;
  #pragma unroll
  for (int mt = 0; mt < 4; mt++) {
    #pragma unroll
    for (int nt = 0; nt < 2; nt++) {
      int col = n0 + wn * 32 + nt * 16 + nl;
      float bias = bp[idx * HD + col];
      #pragma unroll
      for (int v = 0; v < 8; v++) {
        int row = m0 + wm * 64 + mt * 16 + v + 8 * kh;
        float val = acc[mt][nt][v] + bias;
        pout[(size_t)row * HD + col] = fmaxf(val, 0.f);
      }
    }
  }
}

// ---------------------------------------------------------------- LayerNorm + h->bf16
__global__ __launch_bounds__(256) void k_layernorm(
    const float* __restrict__ p, const float* __restrict__ gamma,
    const float* __restrict__ beta, const float* __restrict__ hx,
    const int* __restrict__ cidx, int t,
    unsigned short* __restrict__ lnout, unsigned short* __restrict__ hbf) {
  const int row = blockIdx.x;
  const int tid = threadIdx.x;
  const int idx = cidx[t];
  const float* pr = p + (size_t)row * HD;
  float vals[4], s = 0.f, ss = 0.f;
  #pragma unroll
  for (int j = 0; j < 4; j++) {
    float v = pr[tid + j * 256];
    vals[j] = v; s += v; ss += v * v;
  }
  __shared__ float rs[256], rq[256];
  rs[tid] = s; rq[tid] = ss;
  __syncthreads();
  for (int off = 128; off > 0; off >>= 1) {
    if (tid < off) { rs[tid] += rs[tid + off]; rq[tid] += rq[tid + off]; }
    __syncthreads();
  }
  __shared__ float smu, srstd;
  if (tid == 0) {
    float mu = rs[0] * (1.0f / HD);
    float var = rq[0] * (1.0f / HD) - mu * mu;
    smu = mu; srstd = rsqrtf(var + LNEPS);
  }
  __syncthreads();
  float mu = smu, rstd = srstd;
  const float* hrow = hx + ((size_t)idx * BSZ + row) * HD;
  #pragma unroll
  for (int j = 0; j < 4; j++) {
    int h = tid + j * 256;
    float ln = (vals[j] - mu) * rstd * gamma[idx * HD + h] + beta[idx * HD + h];
    lnout[(size_t)row * HD + h] = f2bf(ln);
    hbf[(size_t)row * HD + h] = f2bf(hrow[h]);
  }
}

// ---------------------------------------------------------------- GEMM 2+3 fused + LSTM
// gates = ln@W_ih[idx]^T + h@W_hh[idx]^T + biases; LSTM update of hx/cx; hn -> bf16
// Workgroup computes tile: 128 rows x 32 cols for ALL FOUR gate blocks.
__global__ __launch_bounds__(256) void k_gates_lstm(
    const unsigned short* __restrict__ ln,   // [B,H] bf16
    const unsigned short* __restrict__ hbf,  // [B,H] bf16
    const unsigned short* __restrict__ Wih,  // [NC,4H,H] bf16
    const unsigned short* __restrict__ Whh,  // [NC,4H,H] bf16
    const float* __restrict__ bih, const float* __restrict__ bhh,
    const int* __restrict__ cidx, int t,
    float* __restrict__ hx, float* __restrict__ cx,
    unsigned short* __restrict__ hnbf) {
  const int idx = cidx[t];
  const unsigned short* Wi = Wih + (size_t)idx * 4 * HD * HD;
  const unsigned short* Wh = Whh + (size_t)idx * 4 * HD * HD;
  const int m0 = blockIdx.y * 128;
  const int n0 = blockIdx.x * 32;
  __shared__ __align__(16) unsigned short sLn[2][128 * LDS_STRIDE];
  __shared__ __align__(16) unsigned short sH [2][128 * LDS_STRIDE];
  __shared__ __align__(16) unsigned short sWi[2][128 * LDS_STRIDE]; // rows: gate*32 + n
  __shared__ __align__(16) unsigned short sWh[2][128 * LDS_STRIDE];
  const int tid = threadIdx.x;
  const int w = tid >> 5, lane = tid & 31;   // wave w owns rows [w*16, w*16+16)
  v8f acc[4][2] = {};                        // [gate][ntile]

  auto prefetch = [&](int buf, int k0) {
    #pragma unroll
    for (int c = tid; c < 512; c += 256) {
      int r = c >> 2, q = c & 3;
      cp16(&sLn[buf][r * LDS_STRIDE + q * 8], ln  + (size_t)(m0 + r) * HD + k0 + q * 8);
      cp16(&sH [buf][r * LDS_STRIDE + q * 8], hbf + (size_t)(m0 + r) * HD + k0 + q * 8);
      size_t wr = (size_t)((r >> 5) * HD + n0 + (r & 31));
      cp16(&sWi[buf][r * LDS_STRIDE + q * 8], Wi + wr * HD + k0 + q * 8);
      cp16(&sWh[buf][r * LDS_STRIDE + q * 8], Wh + wr * HD + k0 + q * 8);
    }
  };

  prefetch(0, 0);
  for (int it = 0; it < NKIT; ++it) {
    wait_async();
    __syncthreads();
    if (it + 1 < NKIT) prefetch((it + 1) & 1, (it + 1) * 32);
    const int cur = it & 1;
    v16bf a_ln = ld_frag(&sLn[cur][(w * 16) * LDS_STRIDE]);
    v16bf a_h  = ld_frag(&sH [cur][(w * 16) * LDS_STRIDE]);
    #pragma unroll
    for (int g = 0; g < 4; g++) {
      #pragma unroll
      for (int nt = 0; nt < 2; nt++) {
        v16bf bi = ld_frag(&sWi[cur][(g * 32 + nt * 16) * LDS_STRIDE]);
        acc[g][nt] = wmma_bf16(a_ln, bi, acc[g][nt]);
        v16bf bh = ld_frag(&sWh[cur][(g * 32 + nt * 16) * LDS_STRIDE]);
        acc[g][nt] = wmma_bf16(a_h, bh, acc[g][nt]);
      }
    }
  }

  const int kh = lane >> 4, nl = lane & 15;
  const size_t cell = (size_t)idx * BSZ * HD;
  #pragma unroll
  for (int nt = 0; nt < 2; nt++) {
    int col = n0 + nt * 16 + nl;
    float bi_i = bih[idx * 4 * HD + 0 * HD + col] + bhh[idx * 4 * HD + 0 * HD + col];
    float bi_f = bih[idx * 4 * HD + 1 * HD + col] + bhh[idx * 4 * HD + 1 * HD + col];
    float bi_g = bih[idx * 4 * HD + 2 * HD + col] + bhh[idx * 4 * HD + 2 * HD + col];
    float bi_o = bih[idx * 4 * HD + 3 * HD + col] + bhh[idx * 4 * HD + 3 * HD + col];
    #pragma unroll
    for (int v = 0; v < 8; v++) {
      int row = m0 + w * 16 + v + 8 * kh;
      size_t o = cell + (size_t)row * HD + col;
      float ig = acc[0][nt][v] + bi_i;
      float fg = acc[1][nt][v] + bi_f;
      float gg = acc[2][nt][v] + bi_g;
      float og = acc[3][nt][v] + bi_o;
      float c_old = cx[o];
      float cn = sigmoidf_(fg) * c_old + sigmoidf_(ig) * tanhf(gg);
      float hn = sigmoidf_(og) * tanhf(cn);
      cx[o] = cn;
      hx[o] = hn;
      hnbf[(size_t)row * HD + col] = f2bf(hn);
    }
  }
}

// ---------------------------------------------------------------- GEMM 4: association + ctx update
// out = tanh(hn @ W_a[idx]^T + b_a[idx]) * sigmoid(gl[idx]);  ctx = 0.8*ctx + 0.2*out
__global__ __launch_bounds__(256) void k_gemm_outctx(
    const unsigned short* __restrict__ hn,   // [B,H] bf16
    const unsigned short* __restrict__ Wa,   // [NC,H,H] bf16
    const float* __restrict__ ba, const float* __restrict__ gl,
    const int* __restrict__ cidx, int t,
    float* __restrict__ ctx) {
  const int idx = cidx[t];
  const unsigned short* W = Wa + (size_t)idx * HD * HD;
  const float gs = sigmoidf_(gl[idx]);
  const int m0 = blockIdx.y * 128;
  const int n0 = blockIdx.x * 128;
  __shared__ __align__(16) unsigned short sA[2][128 * LDS_STRIDE];
  __shared__ __align__(16) unsigned short sB[2][128 * LDS_STRIDE];
  const int tid = threadIdx.x;
  const int w = tid >> 5, lane = tid & 31;
  const int wm = w & 1, wn = w >> 1;
  v8f acc[4][2] = {};

  auto prefetch = [&](int buf, int k0) {
    #pragma unroll
    for (int c = tid; c < 512; c += 256) {
      int r = c >> 2, q = c & 3;
      cp16(&sA[buf][r * LDS_STRIDE + q * 8], hn + (size_t)(m0 + r) * HD + k0 + q * 8);
      cp16(&sB[buf][r * LDS_STRIDE + q * 8], W + (size_t)(n0 + r) * HD + k0 + q * 8);
    }
  };

  prefetch(0, 0);
  for (int it = 0; it < NKIT; ++it) {
    wait_async();
    __syncthreads();
    if (it + 1 < NKIT) prefetch((it + 1) & 1, (it + 1) * 32);
    const int cur = it & 1;
    v16bf bfr[2];
    #pragma unroll
    for (int nt = 0; nt < 2; nt++) bfr[nt] = ld_frag(&sB[cur][(wn * 32 + nt * 16) * LDS_STRIDE]);
    #pragma unroll
    for (int mt = 0; mt < 4; mt++) {
      v16bf a = ld_frag(&sA[cur][(wm * 64 + mt * 16) * LDS_STRIDE]);
      #pragma unroll
      for (int nt = 0; nt < 2; nt++) acc[mt][nt] = wmma_bf16(a, bfr[nt], acc[mt][nt]);
    }
  }

  const int kh = lane >> 4, nl = lane & 15;
  #pragma unroll
  for (int mt = 0; mt < 4; mt++) {
    #pragma unroll
    for (int nt = 0; nt < 2; nt++) {
      int col = n0 + wn * 32 + nt * 16 + nl;
      float bias = ba[idx * HD + col];
      #pragma unroll
      for (int v = 0; v < 8; v++) {
        int row = m0 + wm * 64 + mt * 16 + v + 8 * kh;
        size_t o = (size_t)row * HD + col;
        float out = tanhf(acc[mt][nt][v] + bias) * gs;
        ctx[o] = 0.8f * ctx[o] + 0.2f * out;
      }
    }
  }
}

// ---------------------------------------------------------------- launch
extern "C" void kernel_launch(void* const* d_in, const int* in_sizes, int n_in,
                              void* d_out, int out_size, void* d_ws, size_t ws_size,
                              hipStream_t stream) {
  (void)in_sizes; (void)n_in; (void)out_size; (void)ws_size;
  const float* x     = (const float*)d_in[0];
  const float* W_p   = (const float*)d_in[1];
  const float* b_p   = (const float*)d_in[2];
  const float* gamma = (const float*)d_in[3];
  const float* beta  = (const float*)d_in[4];
  const float* W_ih  = (const float*)d_in[5];
  const float* b_ih  = (const float*)d_in[6];
  const float* W_hh  = (const float*)d_in[7];
  const float* b_hh  = (const float*)d_in[8];
  const float* W_a   = (const float*)d_in[9];
  const float* b_a   = (const float*)d_in[10];
  const float* gl    = (const float*)d_in[11];
  const int*   cidx  = (const int*)d_in[12];
  float* ctx = (float*)d_out;
  char* ws = (char*)d_ws;

  constexpr size_t E_WP  = (size_t)NCELLS * HD * HD;       // 16M
  constexpr size_t E_WIH = (size_t)NCELLS * 4 * HD * HD;   // 64M
  constexpr size_t E_ST  = (size_t)NCELLS * BSZ * HD;      // 32M
  constexpr size_t E_ACT = (size_t)BSZ * HD;               // 2M
  constexpr size_t OFF_WP  = 0;
  constexpr size_t OFF_WIH = OFF_WP  + E_WP  * 2;
  constexpr size_t OFF_WHH = OFF_WIH + E_WIH * 2;
  constexpr size_t OFF_WA  = OFF_WHH + E_WIH * 2;
  constexpr size_t OFF_HX  = OFF_WA  + E_WP  * 2;
  constexpr size_t OFF_CX  = OFF_HX  + E_ST  * 4;
  constexpr size_t OFF_XI  = OFF_CX  + E_ST  * 4;
  constexpr size_t OFF_P   = OFF_XI  + E_ACT * 2;
  constexpr size_t OFF_LN  = OFF_P   + E_ACT * 4;
  constexpr size_t OFF_HB  = OFF_LN  + E_ACT * 2;
  constexpr size_t OFF_HN  = OFF_HB  + E_ACT * 2;

  unsigned short* wp_bf  = (unsigned short*)(ws + OFF_WP);
  unsigned short* wih_bf = (unsigned short*)(ws + OFF_WIH);
  unsigned short* whh_bf = (unsigned short*)(ws + OFF_WHH);
  unsigned short* wa_bf  = (unsigned short*)(ws + OFF_WA);
  float* hx = (float*)(ws + OFF_HX);
  float* cx = (float*)(ws + OFF_CX);
  unsigned short* xi_bf = (unsigned short*)(ws + OFF_XI);
  float* p_f32 = (float*)(ws + OFF_P);
  unsigned short* ln_bf = (unsigned short*)(ws + OFF_LN);
  unsigned short* h_bf  = (unsigned short*)(ws + OFF_HB);
  unsigned short* hn_bf = (unsigned short*)(ws + OFF_HN);

  // init state: ctx = 0, hx = cx = 0 (hx/cx contiguous)
  k_zero_f32<<<(int)(E_ACT / 1024), 256, 0, stream>>>(ctx, (int)E_ACT);
  k_zero_f32<<<(int)(2 * E_ST / 1024), 256, 0, stream>>>(hx, (int)(2 * E_ST));
  // one-time weight conversion f32 -> bf16
  k_cvt_bf16<<<(int)(E_WP  / 1024), 256, 0, stream>>>(W_p,  wp_bf,  (int)E_WP);
  k_cvt_bf16<<<(int)(E_WIH / 1024), 256, 0, stream>>>(W_ih, wih_bf, (int)E_WIH);
  k_cvt_bf16<<<(int)(E_WIH / 1024), 256, 0, stream>>>(W_hh, whh_bf, (int)E_WIH);
  k_cvt_bf16<<<(int)(E_WP  / 1024), 256, 0, stream>>>(W_a,  wa_bf,  (int)E_WP);

  for (int t = 0; t < NACT; ++t) {
    k_xi<<<(int)(E_ACT / 1024), 256, 0, stream>>>(x, ctx, xi_bf, (int)E_ACT);
    k_gemm_percept<<<dim3(HD / 128, BSZ / 128), dim3(256), 0, stream>>>(
        xi_bf, wp_bf, b_p, cidx, t, p_f32);
    k_layernorm<<<BSZ, 256, 0, stream>>>(p_f32, gamma, beta, hx, cidx, t, ln_bf, h_bf);
    k_gates_lstm<<<dim3(HD / 32, BSZ / 128), dim3(256), 0, stream>>>(
        ln_bf, h_bf, wih_bf, whh_bf, b_ih, b_hh, cidx, t, hx, cx, hn_bf);
    k_gemm_outctx<<<dim3(HD / 128, BSZ / 128), dim3(256), 0, stream>>>(
        hn_bf, wa_bf, b_a, gl, cidx, t, ctx);
  }
}
